// CausalAttention_78855599555350
// MI455X (gfx1250) — matrix operand order repaired and checked by reading, not compiled, not run on hardware
//
#include <hip/hip_runtime.h>

// ---------------------------------------------------------------------------
// Causal attention fwd for B=4, S=2048, D=1024 on gfx1250 (MI455X).
// bf16 WMMA everywhere: qkv projection GEMM -> V transpose -> flash attention.
// ---------------------------------------------------------------------------

#define SEQ   2048
#define NB    4
#define DIM   1024
#define MTOT  (NB * SEQ)          // 8192 rows of x flattened

typedef __attribute__((ext_vector_type(8)))  float   v8f;
typedef __attribute__((ext_vector_type(16))) __bf16  v16bf;
typedef __attribute__((ext_vector_type(8)))  __bf16  v8bf;
typedef __attribute__((ext_vector_type(4)))  float   v4f;

__device__ __forceinline__ v16bf cat8(v8bf lo, v8bf hi) {
  return __builtin_shufflevector(lo, hi, 0,1,2,3,4,5,6,7,8,9,10,11,12,13,14,15);
}

__device__ __forceinline__ v16bf wmma_bf16(v16bf a, v16bf b, v8f& c) {
  c = __builtin_amdgcn_wmma_f32_16x16x32_bf16(false, a, false, b, (short)0, c,
                                              false, false);
  return a;
}

// ---------------------------------------------------------------------------
// Kernel 1: C[M,N] = X[M,K] * W[K,N] in bf16 WMMA, f32 inputs converted on the
// fly. BLOCK_M=128, BLOCK_N=64, BLOCK_K=32, 256 threads (8 waves), each wave
// owns a 32x32 patch (2x2 WMMA tiles).  blockIdx.z selects Wq/Wk/Wv.
// ---------------------------------------------------------------------------
__global__ __launch_bounds__(256) void qkv_gemm(
    const float* __restrict__ X,
    const float* __restrict__ Wq, const float* __restrict__ Wk,
    const float* __restrict__ Wv,
    __bf16* __restrict__ Qo, __bf16* __restrict__ Ko, __bf16* __restrict__ Vo) {

  __shared__ __bf16 A_lds[128][40];   // 128x32 tile, stride 40 (80B, 16B mult)
  __shared__ __bf16 B_lds[64][40];    // 64(n) x 32(k) transposed tile

  const int z  = blockIdx.z;
  const float* __restrict__ W = (z == 0) ? Wq : (z == 1) ? Wk : Wv;
  __bf16* __restrict__ O      = (z == 0) ? Qo : (z == 1) ? Ko : Vo;

  const int m0 = blockIdx.x * 128;
  const int n0 = blockIdx.y * 64;

  const int t    = threadIdx.x;
  const int wid  = t >> 5;
  const int lane = t & 31;
  const int l16  = lane & 15;
  const bool hib = lane >= 16;
  const int wm   = wid >> 1;          // 0..3 -> 32-row slice
  const int wn   = wid & 1;           // 0..1 -> 32-col slice

  // LDS fill mapping
  const int am  = t >> 1;             // A: row 0..127
  const int ak  = (t & 1) * 16;       // A: k offset 0/16
  const int bn  = t & 63;             // B: n 0..63
  const int bk  = (t >> 6) * 8;       // B: k offset 0/8/16/24

  const int aoff = hib ? 16 : 0;      // A-frag element offset (contiguous k = 16h + s)
  const int boff = hib ? 16 : 0;      // B-frag element offset (contiguous k = 16h + s)

  v8f acc[4] = {};                    // [tm*2+tn]

  for (int k0 = 0; k0 < DIM; k0 += 32) {
    // ---- fill A tile (f32 -> bf16) ----
    {
      const float* src = X + (size_t)(m0 + am) * DIM + k0 + ak;
      float f[16];
      #pragma unroll
      for (int i = 0; i < 4; i++) {
        v4f v = *(const v4f*)(src + 4 * i);
        f[4*i+0] = v.x; f[4*i+1] = v.y; f[4*i+2] = v.z; f[4*i+3] = v.w;
      }
      v8bf lo, hi;
      #pragma unroll
      for (int i = 0; i < 8; i++) { lo[i] = (__bf16)f[i]; hi[i] = (__bf16)f[i+8]; }
      *(v8bf*)&A_lds[am][ak]     = lo;
      *(v8bf*)&A_lds[am][ak + 8] = hi;
      if (k0 + 32 < DIM) __builtin_prefetch(src + 32, 0, 1);
    }
    // ---- fill B tile transposed: B_lds[n][k] ----
    {
      v8bf wb;
      #pragma unroll
      for (int j = 0; j < 8; j++)
        wb[j] = (__bf16)W[(size_t)(k0 + bk + j) * DIM + n0 + bn];
      *(v8bf*)&B_lds[bn][bk] = wb;
      if (k0 + 32 < DIM) __builtin_prefetch(&W[(size_t)(k0 + 32 + bk) * DIM + n0 + bn], 0, 1);
    }
    __syncthreads();

    // ---- WMMA on the tile ----
    v16bf afrag[2], bfrag[2];
    #pragma unroll
    for (int tm = 0; tm < 2; tm++) {
      const int row = wm * 32 + tm * 16 + l16;
      afrag[tm] = cat8(*(const v8bf*)&A_lds[row][aoff],
                       *(const v8bf*)&A_lds[row][aoff + 8]);
    }
    #pragma unroll
    for (int tn = 0; tn < 2; tn++) {
      const int nrow = wn * 32 + tn * 16 + l16;
      bfrag[tn] = *(const v16bf*)&B_lds[nrow][boff];
    }
    #pragma unroll
    for (int tm = 0; tm < 2; tm++)
      #pragma unroll
      for (int tn = 0; tn < 2; tn++)
        acc[tm * 2 + tn] = __builtin_amdgcn_wmma_f32_16x16x32_bf16(
            false, afrag[tm], false, bfrag[tn], (short)0, acc[tm * 2 + tn],
            false, false);
    __syncthreads();
  }

  // ---- writeback bf16 ----
  #pragma unroll
  for (int tm = 0; tm < 2; tm++)
    #pragma unroll
    for (int tn = 0; tn < 2; tn++) {
      const int col = n0 + wn * 32 + tn * 16 + l16;
      #pragma unroll
      for (int v = 0; v < 8; v++) {
        const int row = m0 + wm * 32 + tm * 16 + v + (hib ? 8 : 0);
        O[(size_t)row * DIM + col] = (__bf16)acc[tm * 2 + tn][v];
      }
    }
}

// ---------------------------------------------------------------------------
// Kernel 2: Vt[d][key] = V[key][d]  (bf16, LDS 64x64 tile transpose)
// ---------------------------------------------------------------------------
__global__ __launch_bounds__(256) void transpose_v(
    const __bf16* __restrict__ V, __bf16* __restrict__ Vt) {
  __shared__ __bf16 tile[64][80];     // stride 160B (16B multiple)
  const int db = blockIdx.x * 64;
  const int kb = blockIdx.y * 64;
  const int t  = threadIdx.x;
  {
    const int key = t >> 2, seg = t & 3;
    const __bf16* src = V + (size_t)(kb + key) * DIM + db + seg * 16;
    *(v8bf*)&tile[key][seg * 16]     = *(const v8bf*)(src);
    *(v8bf*)&tile[key][seg * 16 + 8] = *(const v8bf*)(src + 8);
  }
  __syncthreads();
  {
    const int d = t >> 2, seg = t & 3;
    v8bf lo, hi;
    #pragma unroll
    for (int i = 0; i < 8; i++) lo[i] = tile[seg * 16 + i][d];
    #pragma unroll
    for (int i = 0; i < 8; i++) hi[i] = tile[seg * 16 + 8 + i][d];
    __bf16* dst = Vt + (size_t)(db + d) * MTOT + kb + seg * 16;
    *(v8bf*)dst       = lo;
    *(v8bf*)(dst + 8) = hi;
  }
}

// ---------------------------------------------------------------------------
// Kernel 3: flash attention, one (batch, 32-query block) per workgroup.
// 8 waves. Per 64-key block:
//   stage1: each wave -> one 16x16 score tile, full-D WMMA reduction (L2 feed)
//   stage2: online softmax (8 threads/row, shuffle-xor reductions)
//   stage3: each wave -> (16 q-rows x 256 d-cols) of O via P x V^T WMMAs
// ---------------------------------------------------------------------------
__global__ __launch_bounds__(256) void attn(
    const __bf16* __restrict__ Q, const __bf16* __restrict__ K,
    const __bf16* __restrict__ Vt, float* __restrict__ out) {

  __shared__ float  S_lds[32][68];
  __shared__ __bf16 P_lds[32][80];
  __shared__ float  m_lds[32], l_lds[32], c_lds[32];

  const int q0 = blockIdx.x * 32;
  const int b  = blockIdx.y;
  const int t  = threadIdx.x;
  const int wid  = t >> 5;
  const int lane = t & 31;
  const int l16  = lane & 15;
  const bool hib = lane >= 16;
  const int aoff = hib ? 16 : 0;
  const int boff = hib ? 16 : 0;

  // stage-1 tile coords / stage-3 work coords
  const int tm = wid & 1, tn = wid >> 1;   // score tile (2M x 4N)
  const int qg = wid & 1, dq = wid >> 1;   // output (q half, 256-d chunk)

  if (t < 32) { m_lds[t] = -INFINITY; l_lds[t] = 0.f; }
  __syncthreads();

  v8f acc[16] = {};                        // O accum: 16 n-tiles of 16 cols

  const __bf16* qrow = Q + (size_t)(b * SEQ + q0 + tm * 16 + l16) * DIM;
  const int kb_last = (q0 + 31) >> 6;

  for (int kb = 0; kb <= kb_last; ++kb) {
    const int k0 = kb * 64;

    // ---------------- stage 1: S tile = Q x K^T over full D --------------
    {
      v8f s = {};
      const __bf16* krow = K + (size_t)(b * SEQ + k0 + tn * 16 + l16) * DIM;
      #pragma unroll 4
      for (int d = 0; d < DIM; d += 32) {
        v16bf a  = cat8(*(const v8bf*)(qrow + d + aoff),
                        *(const v8bf*)(qrow + d + aoff + 8));
        v16bf bm = *(const v16bf*)(krow + d + boff);
        s = __builtin_amdgcn_wmma_f32_16x16x32_bf16(false, a, false, bm,
                                                    (short)0, s, false, false);
      }
      const int srb  = tm * 16 + (hib ? 8 : 0);
      const int scol = tn * 16 + l16;
      const int kglb = k0 + scol;
      #pragma unroll
      for (int v = 0; v < 8; v++) {
        const int r = srb + v;
        float val = s[v] * 0.03125f;                 // 1/sqrt(1024)
        if (kglb > q0 + r) val = -INFINITY;          // causal mask
        S_lds[r][scol] = val;
      }
    }
    __syncthreads();

    // ---------------- stage 2: online softmax ----------------------------
    {
      const int r = t >> 3, slot = t & 7;
      float sv[8];
      float pm = -INFINITY;
      #pragma unroll
      for (int i = 0; i < 8; i++) {
        sv[i] = S_lds[r][slot * 8 + i];
        pm = fmaxf(pm, sv[i]);
      }
      pm = fmaxf(pm, __shfl_xor(pm, 1, 32));
      pm = fmaxf(pm, __shfl_xor(pm, 2, 32));
      pm = fmaxf(pm, __shfl_xor(pm, 4, 32));
      const float m_prev = m_lds[r];
      const float m_new  = fmaxf(m_prev, pm);
      float ps = 0.f;
      #pragma unroll
      for (int i = 0; i < 8; i++) {
        const float p = __expf(sv[i] - m_new);
        ps += p;
        P_lds[r][slot * 8 + i] = (__bf16)p;
      }
      ps += __shfl_xor(ps, 1, 32);
      ps += __shfl_xor(ps, 2, 32);
      ps += __shfl_xor(ps, 4, 32);
      if (slot == 0) {
        const float corr = __expf(m_prev - m_new);
        c_lds[r] = corr;
        m_lds[r] = m_new;
        l_lds[r] = l_lds[r] * corr + ps;
      }
    }
    __syncthreads();

    // ---------------- stage 3: O = O*corr + P x V^T ----------------------
    {
      const int rbase = qg * 16 + (hib ? 8 : 0);
      float cr[8];
      #pragma unroll
      for (int v = 0; v < 8; v++) cr[v] = c_lds[rbase + v];
      #pragma unroll
      for (int i = 0; i < 16; i++)
        #pragma unroll
        for (int v = 0; v < 8; v++) acc[i][v] *= cr[v];

      const int prow = qg * 16 + l16;
      #pragma unroll
      for (int ks = 0; ks < 2; ks++) {
        v16bf a = cat8(*(const v8bf*)&P_lds[prow][ks * 32 + aoff],
                       *(const v8bf*)&P_lds[prow][ks * 32 + aoff + 8]);
        const size_t kcol = (size_t)(b * SEQ + k0 + ks * 32 + boff);
        #pragma unroll
        for (int tnn = 0; tnn < 16; tnn++) {
          const int d = dq * 256 + tnn * 16 + l16;
          v16bf bm = *(const v16bf*)(Vt + (size_t)d * MTOT + kcol);
          acc[tnn] = __builtin_amdgcn_wmma_f32_16x16x32_bf16(
              false, a, false, bm, (short)0, acc[tnn], false, false);
        }
      }
    }
    // no end-of-loop barrier needed: next stage-1 only writes S_lds and is
    // followed by a barrier before P/c are rewritten.
  }

  // ---------------- normalize + store ------------------------------------
  const int rbase = qg * 16 + (hib ? 8 : 0);
  float li[8];
  #pragma unroll
  for (int v = 0; v < 8; v++) li[v] = 1.f / l_lds[rbase + v];
  #pragma unroll
  for (int tnn = 0; tnn < 16; tnn++) {
    const int d = dq * 256 + tnn * 16 + l16;
    #pragma unroll
    for (int v = 0; v < 8; v++) {
      const int row = b * SEQ + q0 + rbase + v;
      out[(size_t)row * DIM + d] = acc[tnn][v] * li[v];
    }
  }
}

// ---------------------------------------------------------------------------
extern "C" void kernel_launch(void* const* d_in, const int* in_sizes, int n_in,
                              void* d_out, int out_size, void* d_ws,
                              size_t ws_size, hipStream_t stream) {
  (void)in_sizes; (void)n_in; (void)out_size; (void)ws_size;
  const float* x  = (const float*)d_in[0];
  const float* Wq = (const float*)d_in[1];
  const float* Wk = (const float*)d_in[2];
  const float* Wv = (const float*)d_in[3];

  const size_t MAT = (size_t)MTOT * DIM;          // elements per bf16 matrix
  char* ws = (char*)d_ws;
  __bf16* qws  = (__bf16*)(ws);
  __bf16* kws  = (__bf16*)(ws + 2 * MAT);
  __bf16* vws  = (__bf16*)(ws + 4 * MAT);
  __bf16* vtws = (__bf16*)(ws + 6 * MAT);

  qkv_gemm<<<dim3(MTOT / 128, DIM / 64, 3), 256, 0, stream>>>(
      x, Wq, Wk, Wv, qws, kws, vws);
  transpose_v<<<dim3(DIM / 64, MTOT / 64), 256, 0, stream>>>(vws, vtws);
  attn<<<dim3(SEQ / 32, NB), 256, 0, stream>>>(qws, kws, vtws, (float*)d_out);
}
